// DRIOv2Loss_17712445129444
// MI455X (gfx1250) — compile-verified, matrix operand order checked by reading, not run on hardware
//
#include <hip/hip_runtime.h>
#include <math.h>

// ---------------- problem constants ----------------
#define NROWS   4096
#define DIMS    256
#define TAU_F   10.0f
#define QUANT_F 0.5f
#define MULT_F  0.05f
#define NSAMP   500
#define MAX_EPS 64
#define MAX_ITERS 48
#define LOGW_F  (-8.317766166719343f)   // -log(4096)

typedef __attribute__((ext_vector_type(16))) __bf16       v16bf;
typedef __attribute__((ext_vector_type(8)))  float        v8f;
typedef __attribute__((ext_vector_type(8)))  unsigned int v8u;

// ---------------- helpers ----------------
__device__ __forceinline__ unsigned short f32_bf16_rne(float x) {
  unsigned u = __builtin_bit_cast(unsigned, x);
  unsigned r = u + 0x7FFFu + ((u >> 16) & 1u);
  return (unsigned short)(r >> 16);
}
__device__ __forceinline__ float bf16_hi_f32(unsigned short h) {
  return __builtin_bit_cast(float, ((unsigned)h) << 16);
}

// Assemble a 16-element bf16 WMMA fragment from two contiguous 8-element runs
// (ISA 16-bit A/B layout: per lane, K = khalf..khalf+7 and khalf+16..khalf+23).
__device__ __forceinline__ v16bf load_frag(const unsigned short* base) {
  const uint4 q0 = *(const uint4*)(base);
  const uint4 q1 = *(const uint4*)(base + 16);
  v8u t;
  t[0] = q0.x; t[1] = q0.y; t[2] = q0.z; t[3] = q0.w;
  t[4] = q1.x; t[5] = q1.y; t[6] = q1.z; t[7] = q1.w;
  return __builtin_bit_cast(v16bf, t);
}

// ---------------- small kernels ----------------
__global__ void zero_scal_kernel(float* scal) {
  if (threadIdx.x < 2) scal[threadIdx.x] = 0.0f;
}

__global__ void mse_kernel(const float* __restrict__ Xt, const float* __restrict__ Xp,
                           const int* __restrict__ mask, float* __restrict__ scal) {
  __shared__ float sse[256], snm[256];
  float se = 0.f, nm = 0.f;
  for (int i = blockIdx.x * blockDim.x + threadIdx.x; i < NROWS * DIMS;
       i += gridDim.x * blockDim.x) {
    float mk = (float)mask[i];
    float d  = mk * (Xt[i] - Xp[i]);
    se = fmaf(d, d, se);
    nm += mk;
  }
  sse[threadIdx.x] = se; snm[threadIdx.x] = nm; __syncthreads();
  for (int s = 128; s > 0; s >>= 1) {
    if (threadIdx.x < s) { sse[threadIdx.x] += sse[threadIdx.x + s];
                           snm[threadIdx.x] += snm[threadIdx.x + s]; }
    __syncthreads();
  }
  if (threadIdx.x == 0) { atomicAdd(&scal[0], sse[0]); atomicAdd(&scal[1], snm[0]); }
}

// one-time f32 -> bf16 hi/lo split (bf16x3 GEMM operands)
__global__ void conv_bf16_kernel(const float* __restrict__ X,
                                 unsigned short* __restrict__ H,
                                 unsigned short* __restrict__ L) {
  int i = blockIdx.x * blockDim.x + threadIdx.x;
  if (i >= NROWS * DIMS) return;
  float v = X[i];
  unsigned short h = f32_bf16_rne(v);
  H[i] = h;
  L[i] = f32_bf16_rne(v - bf16_hi_f32(h));
}

__global__ void minmax_kernel(const float* __restrict__ Z, const float* __restrict__ Xi,
                              float* __restrict__ dmin, float* __restrict__ dmax) {
  __shared__ float slo[256], shi[256];
  const int d = blockIdx.x;
  float lo = INFINITY, hi = -INFINITY;
  for (int i = threadIdx.x; i < NROWS; i += blockDim.x) {
    float a = Z [(size_t)i * DIMS + d];
    float b = Xi[(size_t)i * DIMS + d];
    lo = fminf(lo, fminf(a, b));
    hi = fmaxf(hi, fmaxf(a, b));
  }
  slo[threadIdx.x] = lo; shi[threadIdx.x] = hi; __syncthreads();
  for (int s = 128; s > 0; s >>= 1) {
    if (threadIdx.x < s) { slo[threadIdx.x] = fminf(slo[threadIdx.x], slo[threadIdx.x + s]);
                           shi[threadIdx.x] = fmaxf(shi[threadIdx.x], shi[threadIdx.x + s]); }
    __syncthreads();
  }
  if (threadIdx.x == 0) { dmin[d] = slo[0]; dmax[d] = shi[0]; }
}

__global__ void diam_kernel(const float* __restrict__ dmin, const float* __restrict__ dmax,
                            float* __restrict__ scal) {
  __shared__ float s[DIMS];
  const int d = threadIdx.x;
  float r = dmax[d] - dmin[d];
  s[d] = r * r; __syncthreads();
  for (int st = DIMS / 2; st > 0; st >>= 1) {
    if (d < st) s[d] += s[d + st];
    __syncthreads();
  }
  if (d == 0) scal[2] = s[0];   // diameter^2
}

__global__ void sqnorm_kernel(const float* __restrict__ Z, const float* __restrict__ Xi,
                              float* __restrict__ sqz, float* __restrict__ sqy) {
  __shared__ float s[DIMS];
  const int row = blockIdx.x;
  const float* src = (row < NROWS) ? (Z + (size_t)row * DIMS)
                                   : (Xi + (size_t)(row - NROWS) * DIMS);
  float v = src[threadIdx.x];
  s[threadIdx.x] = v * v; __syncthreads();
  for (int st = DIMS / 2; st > 0; st >>= 1) {
    if (threadIdx.x < st) s[threadIdx.x] += s[threadIdx.x + st];
    __syncthreads();
  }
  if (threadIdx.x == 0) {
    if (row < NROWS) sqz[row] = s[0]; else sqy[row - NROWS] = s[0];
  }
}

// deterministic subsample: first NSAMP rows of Z (iid gaussian -> same quantile stats)
__global__ void pairdist_kernel(const float* __restrict__ Z, const float* __restrict__ sq,
                                float* __restrict__ d2) {
  const int t = blockIdx.x * blockDim.x + threadIdx.x;
  if (t >= NSAMP * NSAMP) return;
  const int i = t / NSAMP, j = t - i * NSAMP;
  if (i >= j) { d2[t] = -1.0f; return; }
  const float* zi = Z + (size_t)i * DIMS;
  const float* zj = Z + (size_t)j * DIMS;
  float dot = 0.f;
  for (int k = 0; k < DIMS; ++k) dot = fmaf(zi[k], zj[k], dot);
  float v = 0.5f * (sq[i] + sq[j]) - dot;
  d2[t] = (v > 0.f) ? v : -1.0f;
}

// median-by-bisection + epsilon schedule builder (single block)
__global__ void median_sched_kernel(const float* __restrict__ d2, float* __restrict__ sched,
                                    int* __restrict__ nepsI, float* __restrict__ scal) {
  __shared__ float sred[1024];
  __shared__ int   cred[1024];
  const int tid = threadIdx.x, nb = blockDim.x;
  float mx = 0.f; int cnt = 0;
  for (int t = tid; t < NSAMP * NSAMP; t += nb) {
    float v = d2[t];
    if (v > 0.f) { cnt++; mx = fmaxf(mx, v); }
  }
  sred[tid] = mx; cred[tid] = cnt; __syncthreads();
  for (int s = nb / 2; s > 0; s >>= 1) {
    if (tid < s) { sred[tid] = fmaxf(sred[tid], sred[tid + s]); cred[tid] += cred[tid + s]; }
    __syncthreads();
  }
  const float hi0 = sred[0];
  const float target = QUANT_F * (float)cred[0];
  __syncthreads();
  float lo = 0.f, hi = hi0;
  for (int it = 0; it < 48; ++it) {
    float mid = 0.5f * (lo + hi);
    int c = 0;
    for (int t = tid; t < NSAMP * NSAMP; t += nb) {
      float v = d2[t];
      if (v > 0.f && v <= mid) c++;
    }
    cred[tid] = c; __syncthreads();
    for (int s = nb / 2; s > 0; s >>= 1) {
      if (tid < s) cred[tid] += cred[tid + s];
      __syncthreads();
    }
    int ctot = cred[0]; __syncthreads();
    if ((float)ctot < target) lo = mid; else hi = mid;
  }
  if (tid == 0) {
    float eps = fmaxf(hi * MULT_F, 1e-4f);      // pick_epsilon; note blur^P == eps
    scal[3] = eps;
    float diam2 = scal[2];
    float leps = __logf(eps);
    float v = __logf(diam2);                     // = P*log(diameter)
    const float step = 2.0f * __logf(0.9f);      // P*log(SCALING)
    int idx = 0;
    sched[idx++] = diam2;
    while (v > leps && idx < MAX_EPS - 1) { sched[idx++] = __expf(v); v += step; }
    sched[idx++] = eps;
    *nepsI = idx;
  }
}

// ---------------- cost-matrix GEMM: C = 0.5*(sa_i + sb_j) - A.B^T ----------------
// Pre-split bf16 operands (hi/lo), bf16x3 WMMA, each wave does a 16x64 strip
// (A fragments reused across 4 B tiles): 12 WMMA per 20 b128 loads per k-chunk.
__global__ void cost_gemm_bf16x3(const unsigned short* __restrict__ Ah,
                                 const unsigned short* __restrict__ Al,
                                 const unsigned short* __restrict__ Bh,
                                 const unsigned short* __restrict__ Bl,
                                 const float* __restrict__ sa, const float* __restrict__ sb,
                                 float* __restrict__ C) {
  const int lane = threadIdx.x & 31;
  const int wave = threadIdx.x >> 5;
  const int task = blockIdx.x * 8 + wave;        // 16384 tasks (256 ti x 64 j-groups)
  const int ngrp = NROWS / 64;                   // 64 groups of 4 tiles in j
  const int ti   = task / ngrp;
  const int tg   = task - ti * ngrp;
  const int m     = lane & 15;
  const int khalf = (lane >> 4) << 3;            // 0 or 8 (ISA 16-bit A/B layout)
  const unsigned short* arh = Ah + (size_t)(ti * 16 + m) * DIMS;
  const unsigned short* arl = Al + (size_t)(ti * 16 + m) * DIMS;
  v8f acc[4] = {{}, {}, {}, {}};
  for (int k0 = 0; k0 < DIMS; k0 += 32) {
    const v16bf ah = load_frag(arh + k0 + khalf);
    const v16bf al = load_frag(arl + k0 + khalf);
#pragma unroll
    for (int j = 0; j < 4; ++j) {
      const size_t boff = (size_t)(tg * 64 + j * 16 + m) * DIMS + k0 + khalf;
      const v16bf bh = load_frag(Bh + boff);
      const v16bf bl = load_frag(Bl + boff);
      // bf16x3 split product: hi*hi + hi*lo + lo*hi  (~fp32 accuracy)
      acc[j] = __builtin_amdgcn_wmma_f32_16x16x32_bf16(false, ah, false, bh, (short)0, acc[j], false, false);
      acc[j] = __builtin_amdgcn_wmma_f32_16x16x32_bf16(false, ah, false, bl, (short)0, acc[j], false, false);
      acc[j] = __builtin_amdgcn_wmma_f32_16x16x32_bf16(false, al, false, bh, (short)0, acc[j], false, false);
    }
  }
  const int mbase = (lane >> 4) << 3;
#pragma unroll
  for (int j = 0; j < 4; ++j) {
    const int col = tg * 64 + j * 16 + m;
    const float sbn = sb[col];
#pragma unroll
    for (int r = 0; r < 8; ++r) {
      const int M = mbase + r;
      C[(size_t)(ti * 16 + M) * NROWS + col] = 0.5f * (sa[ti * 16 + M] + sbn) - acc[j][r];
    }
  }
}

// ---------------- row-wise damped softmin over a cost matrix ----------------
// out[row] = -e/(1+e/rho) * logsumexp_j( logw + (b[j] - C[row,j])/e )
// t == -2 : init pass (e = sched[0]),  t == -1 : final pass (e = sched[ne-1]),
// t >= ne : identity (out = prev) so extra fixed-count launches are no-ops.
#define LSE_UPD(VAL)                                                         \
  { float v_ = (VAL);                                                        \
    if (v_ > mmax) { ssum = ssum * __expf(mmax - v_) + 1.0f; mmax = v_; }    \
    else           { ssum += __expf(v_ - mmax); } }

__global__ void softmin_kernel(const float* __restrict__ C, const float* __restrict__ b,
                               const float* __restrict__ prev, float* __restrict__ out,
                               int t, const float* __restrict__ sched,
                               const int* __restrict__ nepsI) {
  __shared__ float sm[256], ss[256];
  const int row = blockIdx.x;
  const int tid = threadIdx.x;
  const int ne  = *nepsI;
  float e;
  if (t == -2)      e = sched[0];
  else if (t == -1) e = sched[ne - 1];
  else {
    if (t >= ne) { if (tid == 0) out[row] = prev[row]; return; }
    e = sched[t];
  }
  const float inv_e = 1.0f / e;
  float mmax = -INFINITY, ssum = 0.0f;
  const float4* Crow = (const float4*)(C + (size_t)row * NROWS);
  const float4* Brow = (const float4*)b;
  for (int j4 = tid; j4 < NROWS / 4; j4 += 256) {
    float4 c4 = Crow[j4];
    float bx = 0.f, by = 0.f, bz = 0.f, bw = 0.f;
    if (Brow) { float4 b4 = Brow[j4]; bx = b4.x; by = b4.y; bz = b4.z; bw = b4.w; }
    LSE_UPD(LOGW_F + (bx - c4.x) * inv_e);
    LSE_UPD(LOGW_F + (by - c4.y) * inv_e);
    LSE_UPD(LOGW_F + (bz - c4.z) * inv_e);
    LSE_UPD(LOGW_F + (bw - c4.w) * inv_e);
  }
  sm[tid] = mmax; ss[tid] = ssum; __syncthreads();
  for (int s = 128; s > 0; s >>= 1) {
    if (tid < s) {
      float m1 = sm[tid], m2 = sm[tid + s];
      float M  = fmaxf(m1, m2);
      ss[tid]  = ss[tid] * __expf(m1 - M) + ss[tid + s] * __expf(m2 - M);
      sm[tid]  = M;
    }
    __syncthreads();
  }
  if (tid == 0) {
    float lse  = sm[0] + __logf(ss[0]);
    float damp = 1.0f / (1.0f + e * (1.0f / TAU_F));
    out[row] = -e * damp * lse;
  }
}

__global__ void avg_pots_kernel(float* __restrict__ a, const float* __restrict__ at, int n) {
  int i = blockIdx.x * blockDim.x + threadIdx.x;
  if (i < n) a[i] = 0.5f * (a[i] + at[i]);
}

__global__ void final_kernel(const float* __restrict__ fa_y, const float* __restrict__ fb_x,
                             const float* __restrict__ fa_x, const float* __restrict__ fb_y,
                             const float* __restrict__ scal, float* __restrict__ out) {
  __shared__ float s1[256], s2[256], s3[256], s4[256];
  const int tid = threadIdx.x;
  float a = 0.f, b = 0.f, c = 0.f, d = 0.f;
  const float ir = 1.0f / TAU_F;
  for (int i = tid; i < NROWS; i += 256) {
    a += __expf(-fa_x[i] * ir);
    b += __expf(-fb_x[i] * ir);
    c += __expf(-fb_y[i] * ir);
    d += __expf(-fa_y[i] * ir);
  }
  s1[tid] = a; s2[tid] = b; s3[tid] = c; s4[tid] = d; __syncthreads();
  for (int s = 128; s > 0; s >>= 1) {
    if (tid < s) { s1[tid] += s1[tid + s]; s2[tid] += s2[tid + s];
                   s3[tid] += s3[tid + s]; s4[tid] += s4[tid + s]; }
    __syncthreads();
  }
  if (tid == 0) {
    float eps = scal[3];                         // eps_target = blur^2 = eps
    float uws = TAU_F + 0.5f * eps;              // UnbalancedWeight forward scale
    float F = uws * (s1[0] - s2[0]) * (1.0f / NROWS);
    float G = uws * (s3[0] - s4[0]) * (1.0f / NROWS);
    float se = scal[0], nm = scal[1];
    float recon = (nm > 0.f) ? (se / fmaxf(nm, 1.0f)) : se;
    out[0] = 0.5f * recon + 0.5f * (F + G);      // ALPHA = 0.5
  }
}

// ---------------- host side ----------------
extern "C" void kernel_launch(void* const* d_in, const int* in_sizes, int n_in,
                              void* d_out, int out_size, void* d_ws, size_t ws_size,
                              hipStream_t stream) {
  (void)in_sizes; (void)n_in; (void)out_size; (void)ws_size;
  const float* Xt   = (const float*)d_in[0];   // X_true
  const float* Xp   = (const float*)d_in[1];   // X_pred
  const float* Xi   = (const float*)d_in[2];   // X_imputed (y)
  const float* Z    = (const float*)d_in[3];   // Z_adversary (x)
  const int*   mask = (const int*)  d_in[4];   // mse_mask

  const size_t MB = 1ull << 20;
  char* w = (char*)d_ws;
  float* Cxy = (float*)(w);                    // 64 MB each, 256 MB total (~L2-resident)
  float* Cyx = (float*)(w +  64 * MB);
  float* Cxx = (float*)(w + 128 * MB);
  float* Cyy = (float*)(w + 192 * MB);
  unsigned short* zh  = (unsigned short*)(w + 256 * MB);   // bf16 hi/lo operands, 2 MB each
  unsigned short* zl  = zh  + NROWS * DIMS;
  unsigned short* xih = zl  + NROWS * DIMS;
  unsigned short* xil = xih + NROWS * DIMS;
  float* aux = (float*)(w + 264 * MB);
  float* sqz   = aux;                          // 4096
  float* sqy   = sqz + NROWS;                  // 4096
  float* pots  = sqy + NROWS;                  // a_y, b_x, a_x, b_y   (4*4096)
  float* potst = pots  + 4 * NROWS;            // at_y, bt_x, at_x, bt_y
  float* potsf = potst + 4 * NROWS;            // fa_y, fb_x, fa_x, fb_y
  float* dmin  = potsf + 4 * NROWS;            // 256
  float* dmax  = dmin + DIMS;                  // 256
  float* sched = dmax + DIMS;                  // 64
  float* scal  = sched + MAX_EPS;              // 16 scalars: se, nm, diam2, eps
  int*   nepsI = (int*)(scal + 16);
  float* d2    = (float*)(nepsI + 16);         // 500*500 pair distances

  float *a_y = pots,   *b_x = pots  + NROWS, *a_x = pots  + 2*NROWS, *b_y = pots  + 3*NROWS;
  float *at_y = potst, *bt_x = potst + NROWS, *at_x = potst + 2*NROWS, *bt_y = potst + 3*NROWS;
  float *fa_y = potsf, *fb_x = potsf + NROWS, *fa_x = potsf + 2*NROWS, *fb_y = potsf + 3*NROWS;

  // ---- scalars / hyperparameters (all on device, graph-capture safe) ----
  zero_scal_kernel<<<1, 32, 0, stream>>>(scal);
  mse_kernel<<<2048, 256, 0, stream>>>(Xt, Xp, mask, scal);
  minmax_kernel<<<DIMS, 256, 0, stream>>>(Z, Xi, dmin, dmax);
  diam_kernel<<<1, DIMS, 0, stream>>>(dmin, dmax, scal);
  sqnorm_kernel<<<2 * NROWS, DIMS, 0, stream>>>(Z, Xi, sqz, sqy);
  pairdist_kernel<<<(NSAMP * NSAMP + 255) / 256, 256, 0, stream>>>(Z, sqz, d2);
  median_sched_kernel<<<1, 1024, 0, stream>>>(d2, sched, nepsI, scal);

  // ---- one-time bf16 hi/lo split of GEMM operands ----
  conv_bf16_kernel<<<(NROWS * DIMS) / 256, 256, 0, stream>>>(Z,  zh,  zl);
  conv_bf16_kernel<<<(NROWS * DIMS) / 256, 256, 0, stream>>>(Xi, xih, xil);

  // ---- cost matrices via bf16x3 WMMA GEMM (materialize once; ~L2-resident) ----
  cost_gemm_bf16x3<<<2048, 256, 0, stream>>>(zh,  zl,  xih, xil, sqz, sqy, Cxy);
  cost_gemm_bf16x3<<<2048, 256, 0, stream>>>(xih, xil, zh,  zl,  sqy, sqz, Cyx);
  cost_gemm_bf16x3<<<2048, 256, 0, stream>>>(zh,  zl,  zh,  zl,  sqz, sqz, Cxx);
  cost_gemm_bf16x3<<<2048, 256, 0, stream>>>(xih, xil, xih, xil, sqy, sqy, Cyy);

  // ---- init pass (t = -2): f = logw only, e = eps_list[0] ----
  softmin_kernel<<<NROWS, 256, 0, stream>>>(Cyx, nullptr, nullptr, a_y, -2, sched, nepsI);
  softmin_kernel<<<NROWS, 256, 0, stream>>>(Cxy, nullptr, nullptr, b_x, -2, sched, nepsI);
  softmin_kernel<<<NROWS, 256, 0, stream>>>(Cxx, nullptr, nullptr, a_x, -2, sched, nepsI);
  softmin_kernel<<<NROWS, 256, 0, stream>>>(Cyy, nullptr, nullptr, b_y, -2, sched, nepsI);

  // ---- fixed-count eps-scaling loop; iterations beyond n_eps are identity ----
  for (int t = 0; t < MAX_ITERS; ++t) {
    softmin_kernel<<<NROWS, 256, 0, stream>>>(Cyx, b_x, a_y, at_y, t, sched, nepsI);
    softmin_kernel<<<NROWS, 256, 0, stream>>>(Cxy, a_y, b_x, bt_x, t, sched, nepsI);
    softmin_kernel<<<NROWS, 256, 0, stream>>>(Cxx, a_x, a_x, at_x, t, sched, nepsI);
    softmin_kernel<<<NROWS, 256, 0, stream>>>(Cyy, b_y, b_y, bt_y, t, sched, nepsI);
    avg_pots_kernel<<<(4 * NROWS) / 256, 256, 0, stream>>>(pots, potst, 4 * NROWS);
  }

  // ---- final differentiable extrapolation at e = eps_list[-1] ----
  softmin_kernel<<<NROWS, 256, 0, stream>>>(Cyx, b_x, a_y, fa_y, -1, sched, nepsI);
  softmin_kernel<<<NROWS, 256, 0, stream>>>(Cxy, a_y, b_x, fb_x, -1, sched, nepsI);
  softmin_kernel<<<NROWS, 256, 0, stream>>>(Cxx, a_x, a_x, fa_x, -1, sched, nepsI);
  softmin_kernel<<<NROWS, 256, 0, stream>>>(Cyy, b_y, b_y, fb_y, -1, sched, nepsI);

  final_kernel<<<1, 256, 0, stream>>>(fa_y, fb_x, fa_x, fb_y, scal, (float*)d_out);
}
